// VectorQuantizer_52209622450485
// MI455X (gfx1250) — compile-verified, matrix operand order checked by reading, not run on hardware
//
#include <hip/hip_runtime.h>
#include <hip/hip_bf16.h>

typedef __attribute__((ext_vector_type(16))) _Float16 v16h;
typedef __attribute__((ext_vector_type(8)))  float    v8f;

#define NUM_EMB   1024
#define EMB_DIM   64
#define BATCH     32
#define HWSZ      1024                     // 32*32
#define N_ROWS    (BATCH * HWSZ)           // 32768
#define N_CHUNKS  (NUM_EMB / 16)           // 64 chunks of 16 codes
#define ZQ_ELEMS  (BATCH * EMB_DIM * HWSZ) // 2097152
#define LOSS_OFF  ZQ_ELEMS
#define IDX_OFF   (ZQ_ELEMS + 1)

// workspace layout (bytes)
#define WS_A_OFF     0        // packed A fragments: 64*2*32*16 halves = 131072 B
#define WS_BIAS_OFF  131072   // 1024 f32 = 4096 B
#define WS_IDX_OFF   135168   // 32768 i32 = 131072 B
#define WS_PART_OFF  266240   // 8192 f32 partial sums = 32768 B

#define SM_A_BYTES    131072
#define SM_BIAS_BYTES 4096
#define SM_TOTAL      (SM_A_BYTES + SM_BIAS_BYTES)   // 135168 B dynamic LDS

// ---------------------------------------------------------------------------
// Kernel 1: pack codebook into WMMA A-fragment layout (f16), per ISA 7.12.2
// A 16x32 f16 map: lane L holds code M=L&15; element j (of v16h) in lane-group
// g=L>>4 corresponds to channel k = 32*h + 8*g + j + (j>=8 ? 8 : 0).
__global__ void vq_pack_codebook(const float* __restrict__ emb,
                                 _Float16* __restrict__ aBuf) {
    int tid = blockIdx.x * 256 + threadIdx.x;   // 65536 threads total
    int j    = tid & 15;
    int lane = (tid >> 4) & 31;
    int h    = (tid >> 9) & 1;
    int c    = tid >> 10;                       // chunk 0..63
    int m    = lane & 15;
    int g    = lane >> 4;
    int k    = 32 * h + 8 * g + j + (j >= 8 ? 8 : 0);
    int code = c * 16 + m;
    aBuf[tid] = (_Float16)emb[code * EMB_DIM + k];
}

// Kernel 2: bias[k] = -0.5 * |e_k|^2 in exact fp32
__global__ void vq_code_bias(const float* __restrict__ emb,
                             float* __restrict__ bias) {
    int k = blockIdx.x * 256 + threadIdx.x;     // 1024 threads
    if (k < NUM_EMB) {
        float s = 0.f;
        #pragma unroll
        for (int c = 0; c < EMB_DIM; ++c) {
            float v = emb[k * EMB_DIM + c];
            s += v * v;
        }
        bias[k] = -0.5f * s;
    }
}

// ---------------------------------------------------------------------------
// Kernel 3: main WMMA scoring + argmax. 8 waves/block, 16 rows per wave.
// Packed codebook fragments + bias are staged once into LDS (132 KB of the
// 320 KB/WGP pool) so the 64-chunk inner loop reads A via ds_load_b128
// instead of re-streaming 128 KB from L2 per wave (8x L2 traffic cut).
// score[m,n] = e_m . z_n  (f16 in, f32 acc)  + bias[m];  argmax over m.
__global__ __launch_bounds__(256) void vq_argmax(
        const float*    __restrict__ z,       // (32,64,32,32) f32
        const _Float16* __restrict__ aBuf,    // packed A fragments (global)
        const float*    __restrict__ biasF,   // -0.5|e|^2 (global)
        int*            __restrict__ wsIdx,   // (32768) int
        float*          __restrict__ out) {   // d_out (indices as float)
    extern __shared__ char smem[];            // 135168 B dynamic LDS

    // ---- cooperative stage: 128 KB A fragments + 4 KB bias -> LDS ----
    {
        const uint4* aG = (const uint4*)aBuf;          // 8192 x 16B
        uint4*       aS = (uint4*)smem;
        #pragma unroll
        for (int i = 0; i < SM_A_BYTES / 16 / 256; ++i)
            aS[threadIdx.x + i * 256] = aG[threadIdx.x + i * 256];
        const float4* bG = (const float4*)biasF;       // 256 x 16B
        float4*       bS = (float4*)(smem + SM_A_BYTES);
        bS[threadIdx.x] = bG[threadIdx.x];
    }

    const int lane = threadIdx.x & 31;
    const int wv   = threadIdx.x >> 5;
    const int tile = blockIdx.x * 8 + wv;    // 2048 tiles of 16 rows
    const int g    = lane >> 4;

    // ---- build B fragment: 16 z-rows (n = lane&15 of tile), f32 -> f16 ----
    const int n0 = tile * 16 + (lane & 15);  // global row
    const int b  = n0 >> 10;
    const int hw = n0 & 1023;
    const float* zb = z + (size_t)b * (EMB_DIM * HWSZ) + hw;
    v16h bfr0, bfr1;
    #pragma unroll
    for (int j = 0; j < 16; ++j) {
        int k0 = 8 * g + j + (j >= 8 ? 8 : 0);       // half 0: ch 0..31
        bfr0[j] = (_Float16)zb[k0 * HWSZ];
        bfr1[j] = (_Float16)zb[(k0 + 32) * HWSZ];    // half 1: ch 32..63
    }

    __syncthreads();                                 // LDS stage complete

    const v16h* aFrag   = (const v16h*)smem;
    const v8f*  biasVec = (const v8f*)(smem + SM_A_BYTES);

    float best    = -3.0e38f;
    int   bestIdx = 0;

    #pragma unroll 2
    for (int c = 0; c < N_CHUNKS; ++c) {
        v16h a0 = aFrag[(c * 2 + 0) * 32 + lane];
        v16h a1 = aFrag[(c * 2 + 1) * 32 + lane];
        v8f acc = {};
        acc = __builtin_amdgcn_wmma_f32_16x16x32_f16(
                  false, a0, false, bfr0, (short)0, acc, false, false);
        acc = __builtin_amdgcn_wmma_f32_16x16x32_f16(
                  false, a1, false, bfr1, (short)0, acc, false, false);
        v8f bv = biasVec[c * 2 + g];                 // codes c*16+8g .. +7
        int base = c * 16 + 8 * g;
        #pragma unroll
        for (int r = 0; r < 8; ++r) {
            float s = acc[r] + bv[r];
            if (s > best) { best = s; bestIdx = base + r; }
        }
    }

    // combine the two lane-groups (same row n, disjoint code sets)
    float ob = __shfl_xor(best, 16, 32);
    int   oi = __shfl_xor(bestIdx, 16, 32);
    if (ob > best || (ob == best && oi < bestIdx)) { best = ob; bestIdx = oi; }

    if (lane < 16) {
        wsIdx[n0] = bestIdx;
        out[IDX_OFF + n0] = (float)bestIdx;          // indices output
    }
}

// ---------------------------------------------------------------------------
// Kernel 4: gather z_q (B,C,H,W) + per-block squared-error partial sums
__global__ __launch_bounds__(256) void vq_gather_loss(
        const float* __restrict__ z,
        const float* __restrict__ emb,
        const int*   __restrict__ wsIdx,
        float*       __restrict__ partials,
        float*       __restrict__ out) {
    __shared__ float red[256];
    int t  = blockIdx.x * 256 + threadIdx.x;   // 2097152 total
    int b  = t >> 16;
    int c  = (t >> 10) & 63;
    int hw = t & 1023;
    int n  = (b << 10) | hw;
    int idx = wsIdx[n];
    float v = emb[idx * EMB_DIM + c];
    out[t] = v;                                 // z_q_ste forward value == z_q
    float d = z[t] - v;
    red[threadIdx.x] = d * d;
    __syncthreads();
    for (int s = 128; s > 0; s >>= 1) {
        if (threadIdx.x < s) red[threadIdx.x] += red[threadIdx.x + s];
        __syncthreads();
    }
    if (threadIdx.x == 0) partials[blockIdx.x] = red[0];
}

// Kernel 5: deterministic final reduction -> loss
__global__ __launch_bounds__(256) void vq_loss_final(
        const float* __restrict__ partials, float* __restrict__ out) {
    __shared__ float red[256];
    float s = 0.f;
    for (int j = 0; j < 32; ++j) s += partials[threadIdx.x + 256 * j];
    red[threadIdx.x] = s;
    __syncthreads();
    for (int st = 128; st > 0; st >>= 1) {
        if (threadIdx.x < st) red[threadIdx.x] += red[threadIdx.x + st];
        __syncthreads();
    }
    if (threadIdx.x == 0)
        out[LOSS_OFF] = 0.25f * red[0] * (1.0f / (float)ZQ_ELEMS);
}

// ---------------------------------------------------------------------------
extern "C" void kernel_launch(void* const* d_in, const int* in_sizes, int n_in,
                              void* d_out, int out_size, void* d_ws, size_t ws_size,
                              hipStream_t stream) {
    const float* z   = (const float*)d_in[0];   // (32,64,32,32)
    const float* emb = (const float*)d_in[1];   // (1024,64)
    float* out = (float*)d_out;

    char* ws = (char*)d_ws;
    _Float16* wsA    = (_Float16*)(ws + WS_A_OFF);
    float*    wsBias = (float*)   (ws + WS_BIAS_OFF);
    int*      wsIdx  = (int*)     (ws + WS_IDX_OFF);
    float*    wsPart = (float*)   (ws + WS_PART_OFF);

    vq_pack_codebook<<<256, 256, 0, stream>>>(emb, wsA);
    vq_code_bias    <<<4,   256, 0, stream>>>(emb, wsBias);
    vq_argmax       <<<256, 256, SM_TOTAL, stream>>>(z, wsA, wsBias, wsIdx, out);
    vq_gather_loss  <<<ZQ_ELEMS / 256, 256, 0, stream>>>(z, emb, wsIdx, wsPart, out);
    vq_loss_final   <<<1,   256, 0, stream>>>(wsPart, out);
}
